// MsConv1dSplitdiv_37744172597317
// MI455X (gfx1250) — compile-verified, matrix operand order checked by reading, not run on hardware
//
#include <hip/hip_runtime.h>
#include <hip/hip_bf16.h>

typedef __attribute__((ext_vector_type(16))) _Float16 v16h;
typedef __attribute__((ext_vector_type(8)))  _Float16 v8h;
typedef __attribute__((ext_vector_type(8)))  float    v8f;

#define NT   64
#define CTOT 256
#define HW   3136   // 56*56
#define PW   58     // padded width
#define PHW  3364   // 58*58
#define TSEG 8

// ---------------------------------------------------------------------------
// Stage 1: general depthwise temporal conv (reads w11..w27), fp32 NCHW ->
// f16 NHWC with 1-px zero border, via LDS transpose.
// Block = 256 threads = 32 padded pixels x 256 channels. 106 tiles/image.
// ---------------------------------------------------------------------------
__global__ __launch_bounds__(256)
void temporal_pack(const float* __restrict__ x,
                   const float* __restrict__ w11, const float* __restrict__ w13,
                   const float* __restrict__ w15, const float* __restrict__ w17,
                   const float* __restrict__ w21, const float* __restrict__ w23,
                   const float* __restrict__ w25, const float* __restrict__ w27,
                   _Float16* __restrict__ y)
{
    __shared__ _Float16 lbuf[32 * 264];          // 32 px rows, stride 264 (pad)

    const int b  = blockIdx.x;                   // 64 * 106 blocks
    const int nt = b / 106;
    const int p0 = (b % 106) * 32;               // padded pixel base (< 3392)
    const int n  = nt / TSEG, t = nt % TSEG;

    // Phase A: pixel-major reads (coalesced over 32 consecutive pixels)
    const int pl = threadIdx.x & 31;
    const int cb = threadIdx.x >> 5;             // 0..7 (wave-uniform)
    const int pp = p0 + pl;
    const int hh = pp / PW - 1, ww = pp % PW - 1;
    const bool interior = (unsigned)hh < 56u && (unsigned)ww < 56u;
    const int psrc = hh * 56 + ww;
#pragma unroll 4
    for (int i = 0; i < 32; ++i) {
        const int c = cb + (i << 3);             // covers 0..255
        const float* wp; int k, cl;
        if      (c < 16)  { wp = w11; k = 1; cl = c;       }
        else if (c < 32)  { wp = w13; k = 3; cl = c - 16;  }
        else if (c < 48)  { wp = w15; k = 5; cl = c - 32;  }
        else if (c < 64)  { wp = w17; k = 7; cl = c - 48;  }
        else if (c < 112) { wp = w21; k = 1; cl = c - 64;  }
        else if (c < 160) { wp = w23; k = 3; cl = c - 112; }
        else if (c < 208) { wp = w25; k = 5; cl = c - 160; }
        else              { wp = w27; k = 7; cl = c - 208; }
        const int pad = (k - 1) >> 1;
        float acc = 0.f;
        if (interior) {
            for (int j = 0; j < k; ++j) {
                const int ts = t + j - pad;
                if ((unsigned)ts < (unsigned)TSEG)
                    acc += wp[cl * k + j] *
                           x[(((size_t)(n * TSEG + ts)) * CTOT + c) * HW + psrc];
            }
        }
        lbuf[pl * 264 + c] = (_Float16)acc;      // border rows -> zeros
    }
    __syncthreads();

    // Phase B: channel-major writes (padded NHWC), 512B per 8 threads
    const int pl2  = threadIdx.x >> 3;           // 0..31
    const int cblk = (threadIdx.x & 7) * 32;
    const int pp2  = p0 + pl2;
    if (pp2 < PHW) {
        const v8h* src = (const v8h*)&lbuf[pl2 * 264 + cblk];
        v8h* dst = (v8h*)&y[((size_t)nt * PHW + pp2) * CTOT + cblk];
        dst[0] = src[0]; dst[1] = src[1]; dst[2] = src[2]; dst[3] = src[3];
    }
}

// ---------------------------------------------------------------------------
// Pack fp32 weights (Cout,Cin,3,3) into f16 B-fragments in CDNA5 lane order.
// Flat layout: Wp[((ntile*nchunks + chunk)*32 + lane)*16 + e]
// K order: kk = r*Cin + ci, kl(e,half) = e + 8*(half + (e>>3))
// ---------------------------------------------------------------------------
__global__ __launch_bounds__(256)
void pack_weights(const float* __restrict__ w2d, _Float16* __restrict__ Wp,
                  int Cin, int Cout)
{
    int idx = blockIdx.x * 256 + threadIdx.x;
    int total = Cout * Cin * 9;
    if (idx >= total) return;
    int e    = idx & 15;
    int lane = (idx >> 4) & 31;
    int fc   = idx >> 9;
    int nchunks = (Cin * 9) >> 5;
    int chunk = fc % nchunks;
    int ntile = fc / nchunks;
    int half  = lane >> 4;
    int kl = e + 8 * (half + (e >> 3));
    int kk = chunk * 32 + kl;
    int ci = kk % Cin;
    int r  = kk / Cin;
    int nc = ntile * 16 + (lane & 15);
    Wp[idx] = (_Float16)w2d[((size_t)nc * Cin + ci) * 9 + r];
}

// ---------------------------------------------------------------------------
// Stage 2: implicit-GEMM 3x3 conv, v_wmma_f32_16x16x32_f16.
// Wave = M(32 px) x N(64 ch). y is padded NHWC f16 (no bounds checks!);
// out is NCHW fp32 via per-wave LDS transpose for coalesced stores.
// ---------------------------------------------------------------------------
template<int CIN, int NCOLS>
__global__ __launch_bounds__(128)
void conv3x3_wmma(const _Float16* __restrict__ y, const _Float16* __restrict__ Wp,
                  float* __restrict__ out, int cinBase, int coutBase)
{
    constexpr int NCH     = CIN / 32;            // chunks per (kh,kw)
    constexpr int NCHUNKS = 9 * NCH;
    __shared__ float ltile[4][32 * 33];          // per-wave 32x32 f32 tile

    const int lane = threadIdx.x & 31;
    const int wave = threadIdx.x >> 5;
    const int half = lane >> 4;
    const int lm   = lane & 15;

    constexpr int PER_IMG = 98 * NCOLS;
    const int task  = blockIdx.x * 4 + wave;
    const int nt    = task / PER_IMG;
    const int rem   = task % PER_IMG;
    const int mpair = rem / NCOLS;
    const int ncol  = rem % NCOLS;

    const int m0 = mpair * 32;
    const int p0 = m0 + lm, p1 = p0 + 16;
    const int pp0 = (p0 / 56 + 1) * PW + (p0 % 56 + 1);   // padded coords
    const int pp1 = (p1 / 56 + 1) * PW + (p1 % 56 + 1);

    const _Float16* __restrict__ base0 =
        y + ((size_t)nt * PHW + pp0) * CTOT + cinBase + 8 * half;
    const _Float16* __restrict__ base1 =
        y + ((size_t)nt * PHW + pp1) * CTOT + cinBase + 8 * half;
    const _Float16* __restrict__ wb =
        Wp + (size_t)(ncol * 4) * NCHUNKS * 512 + lane * 16;

    v8f acc0[4] = {};
    v8f acc1[4] = {};

    static constexpr int DH[9] = {-1,-1,-1, 0,0,0, 1,1,1};
    static constexpr int DW[9] = {-1, 0, 1,-1,0,1,-1,0,1};

#pragma unroll
    for (int r = 0; r < 9; ++r) {
        const int roff = (DH[r] * PW + DW[r]) * CTOT;     // constant per r
        const _Float16* pa0 = base0 + roff;
        const _Float16* pa1 = base1 + roff;
        const _Float16* wr  = wb + (size_t)(r * NCH) * 512;

#pragma unroll
        for (int cb = 0; cb < NCH; ++cb) {
            union { v16h v; v8h h[2]; } a0, a1;
            a0.h[0] = *(const v8h*)(pa0 + cb * 32);
            a0.h[1] = *(const v8h*)(pa0 + cb * 32 + 16);
            a1.h[0] = *(const v8h*)(pa1 + cb * 32);
            a1.h[1] = *(const v8h*)(pa1 + cb * 32 + 16);

#pragma unroll
            for (int j = 0; j < 4; ++j) {
                const v16h bfr =
                    *(const v16h*)(wr + ((size_t)j * NCHUNKS + cb) * 512);
                acc0[j] = __builtin_amdgcn_wmma_f32_16x16x32_f16(
                    false, a0.v, false, bfr, (short)0, acc0[j], false, false);
                acc1[j] = __builtin_amdgcn_wmma_f32_16x16x32_f16(
                    false, a1.v, false, bfr, (short)0, acc1[j], false, false);
            }
        }
    }

    // Store: transpose 32px x 64ch per wave through LDS, 2 passes of 32 ch.
    float* lw = ltile[wave];
    float* __restrict__ ob =
        out + ((size_t)nt * CTOT + coutBase + ncol * 64) * HW + m0;
#pragma unroll
    for (int pass = 0; pass < 2; ++pass) {
        if (pass) __syncthreads();
#pragma unroll
        for (int jj = 0; jj < 2; ++jj) {
            const int j  = pass * 2 + jj;
            const int nl = jj * 16 + lm;         // 0..31 within pass
#pragma unroll
            for (int rr = 0; rr < 8; ++rr) {
                lw[nl * 33 +      rr + 8 * half] = acc0[j][rr];
                lw[nl * 33 + 16 + rr + 8 * half] = acc1[j][rr];
            }
        }
        __syncthreads();
#pragma unroll 8
        for (int cc = 0; cc < 32; ++cc) {
            ob[(size_t)(pass * 32 + cc) * HW + lane] = lw[cc * 33 + lane];
        }
    }
}

// ---------------------------------------------------------------------------
// Launch
// ---------------------------------------------------------------------------
extern "C" void kernel_launch(void* const* d_in, const int* in_sizes, int n_in,
                              void* d_out, int out_size, void* d_ws, size_t ws_size,
                              hipStream_t stream)
{
    const float* x   = (const float*)d_in[0];
    const float* w11 = (const float*)d_in[1];
    const float* w13 = (const float*)d_in[2];
    const float* w15 = (const float*)d_in[3];
    const float* w17 = (const float*)d_in[4];
    const float* w21 = (const float*)d_in[5];
    const float* w23 = (const float*)d_in[6];
    const float* w25 = (const float*)d_in[7];
    const float* w27 = (const float*)d_in[8];
    const float* w2d1 = (const float*)d_in[9];
    const float* w2d2 = (const float*)d_in[10];
    float* out = (float*)d_out;

    const size_t Y_BYTES   = (size_t)NT * PHW * CTOT * sizeof(_Float16); // ~110 MB
    const size_t WP1_BYTES = (size_t)64 * 64 * 9 * sizeof(_Float16);
    char* ws = (char*)d_ws;
    _Float16* y   = (_Float16*)ws;
    _Float16* Wp1 = (_Float16*)(ws + Y_BYTES);
    _Float16* Wp2 = (_Float16*)(ws + Y_BYTES + WP1_BYTES);

    // Stage 1: temporal mix + f16 padded-NHWC pack (64 imgs * 106 tiles)
    temporal_pack<<<6784, 256, 0, stream>>>(x, w11, w13, w15, w17,
                                            w21, w23, w25, w27, y);

    // Weight packing into WMMA B-fragment order
    pack_weights<<<144, 256, 0, stream>>>(w2d1, Wp1, 64, 64);
    pack_weights<<<1296, 256, 0, stream>>>(w2d2, Wp2, 192, 192);

    // Stage 2: WMMA implicit-GEMM convs
    conv3x3_wmma<64, 1><<<1568, 128, 0, stream>>>(y, Wp1, out, 0, 0);
    conv3x3_wmma<192, 3><<<4704, 128, 0, stream>>>(y, Wp2, out, 64, 64);
}
